// HyenaOperator_55551107006706
// MI455X (gfx1250) — matrix operand order repaired and checked by reading, not compiled, not run on hardware
//
#include <hip/hip_runtime.h>
#include <hip/hip_bf16.h>
#include <math.h>

// ---------------- problem constants ----------------
#define B_SZ   2
#define L_SEQ  4096
#define DD     1024
#define ODD    3072          // (ORDER+1)*D
#define ORDER_ 2
#define NFFT   8192          // 2*L
#define FREQ_  10.0f
#define EMB    33

#define BSTRIDE 40           // padded LDS row stride (elements) for 32-wide K tiles

typedef __bf16 bf16_t;
typedef __attribute__((ext_vector_type(16))) __bf16 bf16x16;
typedef __attribute__((ext_vector_type(8)))  __bf16 bf16x8;
typedef __attribute__((ext_vector_type(8)))  float  f32x8;

union Frag32B { uint4 q[2]; bf16x16 v; };

// ---------------- f32 -> bf16 convert ----------------
__global__ void k_f32_to_bf16(const float* __restrict__ in, bf16_t* __restrict__ out, size_t n) {
    size_t i = (size_t)blockIdx.x * blockDim.x + threadIdx.x;
    if (i < n) out[i] = (bf16_t)in[i];
}

// ---------------- GEMM1: res[b,od,l] = sum_k u[b,l,k]*w_in[od,k] + b_in[od] -------------
// block = 8 waves sharing one 64-wide od strip; B tile staged in double-buffered LDS.
__global__ __launch_bounds__(256) void k_gemm1(const bf16_t* __restrict__ uA,
                                               const bf16_t* __restrict__ wB,
                                               const float*  __restrict__ b_in,
                                               float* __restrict__ res) {
    __shared__ bf16_t Bs[2][64 * BSTRIDE];
    const int t    = threadIdx.x;
    const int lane = t & 31;
    const int wave = t >> 5;
    const int m    = lane & 15;        // A row / B,D column index
    const int h    = lane >> 4;        // lane-half
    const int b    = blockIdx.z;
    const int l0   = (blockIdx.x * 8 + wave) * 16;   // M tile over l
    const int od0  = blockIdx.y * 64;                // N group over od (shared by block)

    // staging role: thread t loads 16B chunk (srow, scol..scol+7) of the 64x32 B tile
    const int srow = t >> 2;
    const int scol = (t & 3) * 8;
    const bf16_t* gB = wB + (size_t)(od0 + srow) * DD + scol;

    f32x8 acc[4];
#pragma unroll
    for (int ns = 0; ns < 4; ++ns) {
        float bias = b_in[od0 + ns * 16 + m];
#pragma unroll
        for (int r = 0; r < 8; ++r) acc[ns][r] = bias;
    }

    const bf16_t* ap = uA + (size_t)(b * L_SEQ + l0 + m) * DD;

    // preload first B tile
    {
        uint4 stg = *(const uint4*)gB;
        *(uint4*)&Bs[0][srow * BSTRIDE + scol] = stg;
    }
    __syncthreads();

    for (int k0 = 0; k0 < DD; k0 += 32) {
        const int cur = (k0 >> 5) & 1;
        const bool more = (k0 + 32) < DD;
        uint4 nstg;
        if (more) nstg = *(const uint4*)(gB + k0 + 32);   // prefetch next tile

        bf16x16 afrag;
        {
            bf16x8 lo = *(const bf16x8*)(ap + k0 + h * 8);
            bf16x8 hi = *(const bf16x8*)(ap + k0 + 16 + h * 8);
#pragma unroll
            for (int e = 0; e < 8; ++e) { afrag[e] = lo[e]; afrag[8 + e] = hi[e]; }
        }
#pragma unroll
        for (int ns = 0; ns < 4; ++ns) {
            const int row = (ns * 16 + m) * BSTRIDE + h * 8;
            Frag32B fb;
            fb.q[0] = *(const uint4*)&Bs[cur][row];
            fb.q[1] = *(const uint4*)&Bs[cur][row + 16];
            acc[ns] = __builtin_amdgcn_wmma_f32_16x16x32_bf16(
                false, afrag, false, fb.v, (short)0, acc[ns], false, false);
        }
        if (more) *(uint4*)&Bs[cur ^ 1][srow * BSTRIDE + scol] = nstg;
        __syncthreads();
    }
#pragma unroll
    for (int ns = 0; ns < 4; ++ns) {
        float* op = res + (size_t)b * ODD * L_SEQ
                        + (size_t)(od0 + ns * 16 + m) * L_SEQ + l0 + h * 8;
        *(f32x8*)op = acc[ns];
    }
}

// ---------------- depthwise conv3 along L ----------------
__global__ void k_conv3(const float* __restrict__ in, const float* __restrict__ cw,
                        const float* __restrict__ cb, float* __restrict__ out) {
    size_t i = (size_t)blockIdx.x * blockDim.x + threadIdx.x;
    const size_t total = (size_t)B_SZ * ODD * L_SEQ;
    if (i >= total) return;
    int l = (int)(i & (L_SEQ - 1));
    int ch = (int)((i / L_SEQ) % ODD);
    float xm = (l > 0)          ? in[i - 1] : 0.f;
    float xc =                    in[i];
    float xp = (l < L_SEQ - 1)  ? in[i + 1] : 0.f;
    out[i] = cw[ch * 3 + 0] * xm + cw[ch * 3 + 1] * xc + cw[ch * 3 + 2] * xp + cb[ch];
}

// ---------------- hyena filter: h[(n*D+d)*L + l] ----------------
__global__ __launch_bounds__(256) void k_filter(const float* __restrict__ w1, const float* __restrict__ b1,
                                                const float* __restrict__ w2, const float* __restrict__ b2,
                                                const float* __restrict__ fbias, const float* __restrict__ fdecay,
                                                float* __restrict__ hbuf) {
    __shared__ float s[64];
    const int l = blockIdx.x;
    const int tid = threadIdx.x;
    const float t = (float)l / (float)(L_SEQ - 1);
    if (tid < 64) {
        float a = b1[tid];
#pragma unroll
        for (int e = 0; e < EMB; ++e)
            a += sinf(t * (float)e * FREQ_) * w1[tid * EMB + e];
        s[tid] = sinf(FREQ_ * a);
    }
    __syncthreads();
    for (int j = tid; j < ORDER_ * DD; j += 256) {
        float acc = b2[j];
#pragma unroll 8
        for (int i = 0; i < 64; ++i) acc += s[i] * w2[j * 64 + i];
        float dec = __expf(-__expf(fdecay[j]) * t);
        hbuf[(size_t)j * L_SEQ + l] = (acc * dec + fbias[j]) * (1.0f / (float)L_SEQ);
    }
}

// ---------------- in-LDS 8192-pt radix-2 FFT ----------------
__device__ inline float2 cmul(float2 a, float2 b) {
    return make_float2(a.x * b.x - a.y * b.y, a.x * b.y + a.y * b.x);
}

__device__ void fft_ip(float2* a, float sign) {
    const int tid = threadIdx.x;
    for (int i = tid; i < NFFT; i += 256) {
        int j = (int)(__brev((unsigned)i) >> 19);   // 13-bit reversal
        if (j > i) { float2 t = a[i]; a[i] = a[j]; a[j] = t; }
    }
    __syncthreads();
    for (int s = 1; s <= 13; ++s) {
        const int half = 1 << (s - 1);
        const float base = sign * 6.283185307179586f / (float)(1 << s);
        for (int k = tid; k < NFFT / 2; k += 256) {
            int j  = k & (half - 1);
            int i0 = ((k >> (s - 1)) << s) + j;
            int i1 = i0 + half;
            float sn, cs;
            __sincosf(base * (float)j, &sn, &cs);
            float2 t  = cmul(make_float2(cs, sn), a[i1]);
            float2 u0 = a[i0];
            a[i1] = make_float2(u0.x - t.x, u0.y - t.y);
            a[i0] = make_float2(u0.x + t.x, u0.y + t.y);
        }
        __syncthreads();
    }
}

// one block per (b,d) channel: z = v + i*h, FFT, Hermitian split, multiply, iFFT, gate by x
__global__ __launch_bounds__(256) void k_fftconv(const float* __restrict__ v_in, size_t vin_bstride,
                                                 const float* __restrict__ hp_base,
                                                 const float* __restrict__ x_base, size_t x_bstride,
                                                 float* __restrict__ v_out) {
    __shared__ float2 Z[NFFT];
    __shared__ float2 P[NFFT];
    const int d = blockIdx.x;
    const int b = blockIdx.y;
    const int tid = threadIdx.x;
    const float* vp = v_in + (size_t)b * vin_bstride + (size_t)d * L_SEQ;
    const float* hp = hp_base + (size_t)d * L_SEQ;

    for (int i = tid; i < NFFT; i += 256) {
        float vr = (i < L_SEQ) ? vp[i] : 0.f;
        float hr = (i < L_SEQ) ? hp[i] : 0.f;
        Z[i] = make_float2(vr, hr);
    }
    __syncthreads();
    fft_ip(Z, -1.f);
    for (int k = tid; k < NFFT; k += 256) {
        float2 Zk = Z[k];
        float2 Zm = Z[(NFFT - k) & (NFFT - 1)];
        float2 V = make_float2(0.5f * (Zk.x + Zm.x), 0.5f * (Zk.y - Zm.y));
        float2 H = make_float2(0.5f * (Zk.y + Zm.y), 0.5f * (Zm.x - Zk.x));
        P[k] = cmul(V, H);
    }
    __syncthreads();
    fft_ip(P, +1.f);
    const float* xp = x_base + (size_t)b * x_bstride + (size_t)d * L_SEQ;
    const float invn = 1.0f / (float)NFFT;
    for (int i = tid; i < L_SEQ; i += 256)
        v_out[((size_t)b * DD + d) * L_SEQ + i] = P[i].x * invn * xp[i];
}

// ---------------- tiled transpose+convert: v(B,D,L) f32 -> vT(B,L,D) bf16 ----------------
__global__ __launch_bounds__(256) void k_transpose_cvt(const float* __restrict__ v,
                                                       bf16_t* __restrict__ vT) {
    __shared__ float tile[32][33];
    const int b = blockIdx.z;
    const int l0 = blockIdx.x * 32;
    const int d0 = blockIdx.y * 32;
    const int x = threadIdx.x;   // 0..31
    const int y = threadIdx.y;   // 0..7
#pragma unroll
    for (int i = 0; i < 4; ++i) {
        int dl = y + i * 8;
        tile[dl][x] = v[((size_t)b * DD + d0 + dl) * L_SEQ + l0 + x];
    }
    __syncthreads();
#pragma unroll
    for (int i = 0; i < 4; ++i) {
        int ll = y + i * 8;
        vT[((size_t)b * L_SEQ + l0 + ll) * DD + d0 + x] = (bf16_t)tile[x][ll];
    }
}

// ---------------- GEMM2: out[b,l,j] = sum_k w_out[j,k]*vT[b,l,k] + b_out[j] -------------
// block = 8 waves sharing one 64-wide l strip; B tile (vT rows) staged in LDS.
__global__ __launch_bounds__(256) void k_gemm2(const bf16_t* __restrict__ wA,
                                               const bf16_t* __restrict__ vT,
                                               const float*  __restrict__ b_out,
                                               float* __restrict__ out) {
    __shared__ bf16_t Bs[2][64 * BSTRIDE];
    const int t    = threadIdx.x;
    const int lane = t & 31;
    const int wave = t >> 5;
    const int m    = lane & 15;
    const int h    = lane >> 4;
    const int b    = blockIdx.z;
    const int dout0 = (blockIdx.x * 8 + wave) * 16;  // M tile over dout
    const int l0    = blockIdx.y * 64;               // N group over l (shared by block)

    const int srow = t >> 2;
    const int scol = (t & 3) * 8;
    const bf16_t* gB = vT + (size_t)(b * L_SEQ + l0 + srow) * DD + scol;

    f32x8 acc[4];
#pragma unroll
    for (int r = 0; r < 8; ++r) {
        float bias = b_out[dout0 + h * 8 + r];
#pragma unroll
        for (int ns = 0; ns < 4; ++ns) acc[ns][r] = bias;
    }

    const bf16_t* ap = wA + (size_t)(dout0 + m) * DD;

    {
        uint4 stg = *(const uint4*)gB;
        *(uint4*)&Bs[0][srow * BSTRIDE + scol] = stg;
    }
    __syncthreads();

    for (int k0 = 0; k0 < DD; k0 += 32) {
        const int cur = (k0 >> 5) & 1;
        const bool more = (k0 + 32) < DD;
        uint4 nstg;
        if (more) nstg = *(const uint4*)(gB + k0 + 32);

        bf16x16 afrag;
        {
            bf16x8 lo = *(const bf16x8*)(ap + k0 + h * 8);
            bf16x8 hi = *(const bf16x8*)(ap + k0 + 16 + h * 8);
#pragma unroll
            for (int e = 0; e < 8; ++e) { afrag[e] = lo[e]; afrag[8 + e] = hi[e]; }
        }
#pragma unroll
        for (int ns = 0; ns < 4; ++ns) {
            const int row = (ns * 16 + m) * BSTRIDE + h * 8;
            Frag32B fb;
            fb.q[0] = *(const uint4*)&Bs[cur][row];
            fb.q[1] = *(const uint4*)&Bs[cur][row + 16];
            acc[ns] = __builtin_amdgcn_wmma_f32_16x16x32_bf16(
                false, afrag, false, fb.v, (short)0, acc[ns], false, false);
        }
        if (more) *(uint4*)&Bs[cur ^ 1][srow * BSTRIDE + scol] = nstg;
        __syncthreads();
    }
#pragma unroll
    for (int ns = 0; ns < 4; ++ns) {
        float* op = out + (size_t)(b * L_SEQ + l0 + ns * 16 + m) * DD + dout0 + h * 8;
        *(f32x8*)op = acc[ns];
    }
}

// ---------------- host-side orchestration ----------------
extern "C" void kernel_launch(void* const* d_in, const int* in_sizes, int n_in,
                              void* d_out, int out_size, void* d_ws, size_t ws_size,
                              hipStream_t stream) {
    (void)in_sizes; (void)n_in; (void)out_size; (void)ws_size;
    const float* u       = (const float*)d_in[0];
    const float* w_in    = (const float*)d_in[1];
    const float* b_in    = (const float*)d_in[2];
    const float* conv_w  = (const float*)d_in[3];
    const float* conv_b  = (const float*)d_in[4];
    const float* w1      = (const float*)d_in[5];
    const float* b1      = (const float*)d_in[6];
    const float* w2      = (const float*)d_in[7];
    const float* b2      = (const float*)d_in[8];
    const float* fbias   = (const float*)d_in[9];
    const float* fdecay  = (const float*)d_in[10];
    const float* w_out   = (const float*)d_in[11];
    const float* b_out   = (const float*)d_in[12];
    float* out = (float*)d_out;

    const size_t SZ_U    = (size_t)B_SZ * L_SEQ * DD;   // 8388608
    const size_t SZ_WIN  = (size_t)ODD * DD;            // 3145728
    const size_t SZ_WOUT = (size_t)DD * DD;             // 1048576
    const size_t SZ_RES  = (size_t)B_SZ * ODD * L_SEQ;  // 25165824
    const size_t SZ_V    = (size_t)B_SZ * DD * L_SEQ;   // 8388608

    char* ws = (char*)d_ws;
    bf16_t* u_bf  = (bf16_t*)(ws);
    bf16_t* wi_bf = (bf16_t*)(ws + SZ_U * 2);
    bf16_t* wo_bf = (bf16_t*)(ws + SZ_U * 2 + SZ_WIN * 2);
    size_t o_res  = SZ_U * 2 + SZ_WIN * 2 + SZ_WOUT * 2;
    float*  res   = (float*)(ws + o_res);
    float*  res_c = (float*)(ws + o_res + SZ_RES * 4);
    float*  h_ws  = (float*)(ws + o_res + 2 * SZ_RES * 4);
    // res region is dead after conv3: alias v and vT into it
    float*  v_ws  = (float*)(ws + o_res);
    bf16_t* vT_bf = (bf16_t*)(ws + o_res + SZ_V * 4);

    // 1) bf16 conversions
    k_f32_to_bf16<<<(unsigned)((SZ_U    + 255) / 256), 256, 0, stream>>>(u,     u_bf,  SZ_U);
    k_f32_to_bf16<<<(unsigned)((SZ_WIN  + 255) / 256), 256, 0, stream>>>(w_in,  wi_bf, SZ_WIN);
    k_f32_to_bf16<<<(unsigned)((SZ_WOUT + 255) / 256), 256, 0, stream>>>(w_out, wo_bf, SZ_WOUT);

    // 2) GEMM1 (WMMA bf16, LDS-staged B): res (B, OD, L)
    k_gemm1<<<dim3(L_SEQ / 16 / 8, ODD / 64, B_SZ), 256, 0, stream>>>(u_bf, wi_bf, b_in, res);

    // 3) depthwise conv3 -> res_c
    k_conv3<<<(unsigned)((SZ_RES + 255) / 256), 256, 0, stream>>>(res, conv_w, conv_b, res_c);

    // 4) hyena filter -> h_ws (ORDER*D, L)
    k_filter<<<L_SEQ, 256, 0, stream>>>(w1, b1, w2, b2, fbias, fdecay, h_ws);

    // 5) FFT conv rounds (in-LDS 8192-pt FFT per channel)
    // round 0: v from res_c rows [2D..3D), x from rows [0..D)
    k_fftconv<<<dim3(DD, B_SZ), 256, 0, stream>>>(
        res_c + (size_t)2 * DD * L_SEQ, (size_t)ODD * L_SEQ,
        h_ws,
        res_c, (size_t)ODD * L_SEQ,
        v_ws);
    // round 1: v from v_ws, x from rows [D..2D)
    k_fftconv<<<dim3(DD, B_SZ), 256, 0, stream>>>(
        v_ws, (size_t)DD * L_SEQ,
        h_ws + (size_t)DD * L_SEQ,
        res_c + (size_t)DD * L_SEQ, (size_t)ODD * L_SEQ,
        v_ws);

    // 6) transpose + convert v -> vT bf16 (B, L, D)
    k_transpose_cvt<<<dim3(L_SEQ / 32, DD / 32, B_SZ), dim3(32, 8), 0, stream>>>(v_ws, vT_bf);

    // 7) GEMM2 (WMMA bf16, LDS-staged B): out (B, L, D)
    k_gemm2<<<dim3(DD / 16 / 8, L_SEQ / 64, B_SZ), 256, 0, stream>>>(wo_bf, vT_bf, b_out, out);
}
